// ScaleGraphBlock_72232759984266
// MI455X (gfx1250) — compile-verified, matrix operand order checked by reading, not run on hardware
//
#include <hip/hip_runtime.h>
#include <hip/hip_bf16.h>
#include <math.h>

// ---------------------------------------------------------------------------
// MI455X / gfx1250: single GAT branch (K=2 branches identical), all GEMMs on
// v_wmma_f32_16x16x32_bf16 with f32 accumulation.
// GEMM: 128x128 block tile, 8 waves x (32x64) subtiles, double-buffered LDS
// (one barrier per K-step), 8 WMMA per wave per K-step.
// ---------------------------------------------------------------------------

typedef __bf16 bf16_t;
typedef __attribute__((ext_vector_type(16))) __bf16 v16bf;
typedef __attribute__((ext_vector_type(8)))  float  v8f;
typedef __attribute__((ext_vector_type(2)))  unsigned int u32x2;

#define B_   16
#define T_   512
#define N_   128
#define SC_  64
#define L_   8           // T/SC
#define G_   128         // B*L graph batches
#define DFF_ 2048

#define LDSP 40          // LDS row pitch in bf16 (32 data + 8 pad, conflict-free)
#define TSZ  (128 * LDSP)

__device__ __forceinline__ v8f zero8() {
  v8f z;
#pragma unroll
  for (int i = 0; i < 8; ++i) z[i] = 0.0f;
  return z;
}

typedef union { v16bf v; u32x2 u[4]; } Frag;

// 16x32 bf16 fragment from an LDS tile [rows][LDSP], per the CDNA5 16-bit
// layout: lane l holds row (base + l%16); VGPRs 0-3 = K k0..k0+7,
// VGPRs 4-7 = K 16+k0..16+k0+7, k0 = 8*(l/16).
__device__ __forceinline__ v16bf load_frag(const bf16_t* lds, int rowBase, int ml, int k0) {
  const bf16_t* r = lds + (rowBase + ml) * LDSP;
  Frag f;
  const u32x2* p0 = (const u32x2*)(r + k0);
  const u32x2* p1 = (const u32x2*)(r + 16 + k0);
  f.u[0] = p0[0]; f.u[1] = p0[1];
  f.u[2] = p1[0]; f.u[3] = p1[1];
  return f.v;
}

// ---------------------------------------------------------------------------
// Batched GEMM: per graph-batch g, C[128 x Ntot] = A[128 x KTOT] @ B + bias.
// B_ROWMAJOR=false: B given as [Ntot][KTOT] row-major (== torch [out][in]).
// B_ROWMAJOR=true : B given as [KTOT][Ntot] row-major (transposed into LDS).
// ---------------------------------------------------------------------------
template <int KTOT, bool B_ROWMAJOR, bool BATCHED_B, bool A_LOCAL_G,
          bool OUT_LOCAL_G, bool OUT_BF16, bool RELU>
__global__ __launch_bounds__(256) void gemm_wmma_kernel(
    const bf16_t* __restrict__ A, const bf16_t* __restrict__ Bm,
    const float* __restrict__ bias, void* __restrict__ outp,
    int Ntot, int gBase) {
  __shared__ bf16_t lA[2 * TSZ];   // 2 x (128 rows x 32 k)
  __shared__ bf16_t lB[2 * TSZ];   // 2 x (128 n   x 32 k)

  const int by    = blockIdx.y;
  const int gGlob = gBase + by;
  const int nTile = blockIdx.x * 128;
  const int tid   = threadIdx.x;
  const int lane  = tid & 31;
  const int wave  = tid >> 5;

  const bf16_t* Ag = A + (size_t)(A_LOCAL_G ? by : gGlob) * (size_t)128 * KTOT;
  const bf16_t* Bg = Bm + (BATCHED_B ? (size_t)gGlob * (size_t)Ntot * KTOT : 0);
  const size_t outOff = (size_t)(OUT_LOCAL_G ? by : gGlob) * (size_t)128 * Ntot;

  const int wm = (wave & 3) * 32;   // 4 wave-rows of 32
  const int wn = (wave >> 2) * 64;  // 2 wave-cols of 64
  const int ml = lane & 15;
  const int hi = lane >> 4;
  const int k0 = hi * 8;

  // staging coordinates (shared by A and col-major B: 128 rows x 32 k)
  const int sRow = tid >> 1;
  const int sCol = (tid & 1) * 16;

  v8f acc[2][4];
#pragma unroll
  for (int i = 0; i < 2; ++i)
#pragma unroll
    for (int j = 0; j < 4; ++j) acc[i][j] = zero8();

  constexpr int NK = KTOT / 32;

  // ---- stage one K-slab into LDS buffer `buf` ----
  auto stage = [&](int buf, int kb) {
    {
      const u32x2* src = (const u32x2*)(Ag + (size_t)sRow * KTOT + kb + sCol);
      u32x2* dst = (u32x2*)(lA + buf * TSZ + sRow * LDSP + sCol);
      dst[0] = src[0]; dst[1] = src[1]; dst[2] = src[2]; dst[3] = src[3];
    }
    if (B_ROWMAJOR) {
      // B is [K][Ntot] row-major; transpose 32x128 tile into LDS [n][k]
      int kr = tid >> 3;             // 32 k-rows
      int n0 = (tid & 7) * 16;       // 16 n per thread
      const bf16_t* src = Bg + (size_t)(kb + kr) * Ntot + nTile + n0;
      bf16_t* dst = lB + buf * TSZ;
#pragma unroll
      for (int i = 0; i < 16; ++i) dst[(n0 + i) * LDSP + kr] = src[i];
    } else {
      const u32x2* src = (const u32x2*)(Bg + (size_t)(nTile + sRow) * KTOT + kb + sCol);
      u32x2* dst = (u32x2*)(lB + buf * TSZ + sRow * LDSP + sCol);
      dst[0] = src[0]; dst[1] = src[1]; dst[2] = src[2]; dst[3] = src[3];
    }
  };

  stage(0, 0);

  for (int it = 0; it < NK; ++it) {
    __syncthreads();                 // buffer it%2 ready; (it+1)%2 free
    if (it + 1 < NK) stage((it + 1) & 1, (it + 1) * 32);
    if (it + 2 < NK)                 // pull the K-slab after next toward L2/L0
      __builtin_prefetch(Ag + (size_t)sRow * KTOT + (it + 2) * 32, 0, 1);

    const bf16_t* cA = lA + (it & 1) * TSZ;
    const bf16_t* cB = lB + (it & 1) * TSZ;
    v16bf a0 = load_frag(cA, wm,      ml, k0);
    v16bf a1 = load_frag(cA, wm + 16, ml, k0);
    v16bf b0 = load_frag(cB, wn,      ml, k0);
    v16bf b1 = load_frag(cB, wn + 16, ml, k0);
    v16bf b2 = load_frag(cB, wn + 32, ml, k0);
    v16bf b3 = load_frag(cB, wn + 48, ml, k0);

    acc[0][0] = __builtin_amdgcn_wmma_f32_16x16x32_bf16(false, a0, false, b0, (short)0, acc[0][0], false, false);
    acc[0][1] = __builtin_amdgcn_wmma_f32_16x16x32_bf16(false, a0, false, b1, (short)0, acc[0][1], false, false);
    acc[0][2] = __builtin_amdgcn_wmma_f32_16x16x32_bf16(false, a0, false, b2, (short)0, acc[0][2], false, false);
    acc[0][3] = __builtin_amdgcn_wmma_f32_16x16x32_bf16(false, a0, false, b3, (short)0, acc[0][3], false, false);
    acc[1][0] = __builtin_amdgcn_wmma_f32_16x16x32_bf16(false, a1, false, b0, (short)0, acc[1][0], false, false);
    acc[1][1] = __builtin_amdgcn_wmma_f32_16x16x32_bf16(false, a1, false, b1, (short)0, acc[1][1], false, false);
    acc[1][2] = __builtin_amdgcn_wmma_f32_16x16x32_bf16(false, a1, false, b2, (short)0, acc[1][2], false, false);
    acc[1][3] = __builtin_amdgcn_wmma_f32_16x16x32_bf16(false, a1, false, b3, (short)0, acc[1][3], false, false);
  }

  // ---- epilogue: C vgpr v -> row (v + 8*hi), col = lane%16 ----
  float*  Of = OUT_BF16 ? nullptr : ((float*)outp + outOff);
  bf16_t* Ob = OUT_BF16 ? ((bf16_t*)outp + outOff) : nullptr;

#pragma unroll
  for (int j = 0; j < 4; ++j) {
    int col = nTile + wn + 16 * j + ml;
    float bv = bias ? bias[col] : 0.0f;
#pragma unroll
    for (int i = 0; i < 2; ++i) {
      int rBase = wm + 16 * i + hi * 8;
#pragma unroll
      for (int v = 0; v < 8; ++v) {
        float q = acc[i][j][v] + bv;
        if (RELU) q = fmaxf(q, 0.0f);
        if (OUT_BF16) Ob[(size_t)(rBase + v) * Ntot + col] = (bf16_t)q;
        else          Of[(size_t)(rBase + v) * Ntot + col] = q;
      }
    }
  }
}

// ---------------------------------------------------------------------------
// Elementwise / reduction kernels
// ---------------------------------------------------------------------------
__global__ void f2bf_kernel(const float* __restrict__ s, bf16_t* __restrict__ d, int n) {
  int i = blockIdx.x * 256 + threadIdx.x;
  if (i < n) d[i] = (bf16_t)s[i];
}

// gat_w[:, :64] -> bf16 [512][64]
__global__ void w1_pack_kernel(const float* __restrict__ gw, bf16_t* __restrict__ w1) {
  int i = blockIdx.x * 256 + threadIdx.x;
  if (i < 512 * 64) {
    int r = i >> 6, c = i & 63;
    w1[i] = (bf16_t)gw[r * 512 + c];
  }
}

// h64[g][n][k] = x[b, l*64+k, n]  (g = b*8 + l), bf16 [128][128][64]
__global__ void h64_pack_kernel(const float* __restrict__ x, bf16_t* __restrict__ h64) {
  int i = blockIdx.x * 256 + threadIdx.x;
  if (i < G_ * N_ * SC_) {
    int g = i >> 13;            // /(128*64)
    int r = i & 8191;
    int n = r >> 6, k = r & 63;
    int b = g >> 3, l = g & 7;
    h64[i] = (bf16_t)x[((size_t)(b * 512 + l * 64 + k)) * 128 + n];
  }
}

// causal masked softmax(scale*s) followed by a second (unmasked) softmax
__global__ __launch_bounds__(128) void softmax2_kernel(
    const float* __restrict__ S, bf16_t* __restrict__ att) {
  __shared__ float red[128];
  int row = blockIdx.x;            // g*128 + n
  int n = row & 127;
  int m = threadIdx.x;
  float s = S[(size_t)row * 128 + m];
  float v1 = (m <= n) ? s * 0.044194173824159216f : -INFINITY;  // 1/sqrt(512)

  red[m] = v1; __syncthreads();
  for (int o = 64; o > 0; o >>= 1) { if (m < o) red[m] = fmaxf(red[m], red[m + o]); __syncthreads(); }
  float mx = red[0]; __syncthreads();
  float e1 = __expf(v1 - mx);
  red[m] = e1; __syncthreads();
  for (int o = 64; o > 0; o >>= 1) { if (m < o) red[m] += red[m + o]; __syncthreads(); }
  float p1 = e1 / red[0]; __syncthreads();

  // second softmax over post-softmax values (masked slots contribute exp(0))
  red[m] = p1; __syncthreads();
  for (int o = 64; o > 0; o >>= 1) { if (m < o) red[m] = fmaxf(red[m], red[m + o]); __syncthreads(); }
  float mx2 = red[0]; __syncthreads();
  float e2 = __expf(p1 - mx2);
  red[m] = e2; __syncthreads();
  for (int o = 64; o > 0; o >>= 1) { if (m < o) red[m] += red[m + o]; __syncthreads(); }
  att[(size_t)row * 128 + m] = (bf16_t)(e2 / red[0]);
}

// x = LN(h_prime + h) over T=512; h[t<64] comes straight from x-input
__global__ __launch_bounds__(256) void ln1_kernel(
    const float* __restrict__ hp, const float* __restrict__ x, bf16_t* __restrict__ xbf) {
  __shared__ float red[256];
  int row = blockIdx.x;            // g*128 + n
  int g = row >> 7, n = row & 127;
  int b = g >> 3, l = g & 7;
  int t = threadIdx.x;
  size_t base = (size_t)row * 512;
  float v0 = hp[base + t];
  float v1 = hp[base + t + 256];
  if (t < 64) v0 += x[((size_t)(b * 512 + l * 64 + t)) * 128 + n];
  red[t] = v0 + v1; __syncthreads();
  for (int o = 128; o > 0; o >>= 1) { if (t < o) red[t] += red[t + o]; __syncthreads(); }
  float mean = red[0] * (1.0f / 512.0f); __syncthreads();
  float d0 = v0 - mean, d1 = v1 - mean;
  red[t] = d0 * d0 + d1 * d1; __syncthreads();
  for (int o = 128; o > 0; o >>= 1) { if (t < o) red[t] += red[t + o]; __syncthreads(); }
  float rstd = rsqrtf(red[0] * (1.0f / 512.0f) + 1e-5f);
  xbf[base + t]       = (bf16_t)(d0 * rstd);
  xbf[base + t + 256] = (bf16_t)(d1 * rstd);
}

// out = LN(x + y) over T=512; store only t<64 (the unpad slice)
__global__ __launch_bounds__(256) void ln2_kernel(
    const bf16_t* __restrict__ xbf, const float* __restrict__ y, float* __restrict__ gout) {
  __shared__ float red[256];
  int row = blockIdx.x;
  int t = threadIdx.x;
  size_t base = (size_t)row * 512;
  float v0 = (float)xbf[base + t]       + y[base + t];
  float v1 = (float)xbf[base + t + 256] + y[base + t + 256];
  red[t] = v0 + v1; __syncthreads();
  for (int o = 128; o > 0; o >>= 1) { if (t < o) red[t] += red[t + o]; __syncthreads(); }
  float mean = red[0] * (1.0f / 512.0f); __syncthreads();
  float d0 = v0 - mean;
  float d1 = v1 - mean;
  red[t] = d0 * d0 + d1 * d1; __syncthreads();
  for (int o = 128; o > 0; o >>= 1) { if (t < o) red[t] += red[t + o]; __syncthreads(); }
  float rstd = rsqrtf(red[0] * (1.0f / 512.0f) + 1e-5f);
  if (t < 64) gout[(size_t)row * 64 + t] = d0 * rstd;
}

// final: out[b,s,:] = LN_over_N( gout[g, :, s%64] + x[b,s,:] )
__global__ __launch_bounds__(128) void final_ln_kernel(
    const float* __restrict__ gout, const float* __restrict__ x, float* __restrict__ out) {
  __shared__ float red[128];
  int bs = blockIdx.x;             // b*512 + s
  int b = bs >> 9, s = bs & 511;
  int g = (b << 3) + (s >> 6), t = s & 63;
  int c = threadIdx.x;
  float v = gout[((size_t)(g * 128 + c)) * 64 + t] + x[(size_t)bs * 128 + c];
  red[c] = v; __syncthreads();
  for (int o = 64; o > 0; o >>= 1) { if (c < o) red[c] += red[c + o]; __syncthreads(); }
  float mean = red[0] * (1.0f / 128.0f); __syncthreads();
  float d = v - mean;
  red[c] = d * d; __syncthreads();
  for (int o = 64; o > 0; o >>= 1) { if (c < o) red[c] += red[c + o]; __syncthreads(); }
  float rstd = rsqrtf(red[0] * (1.0f / 128.0f) + 1e-5f);
  out[(size_t)bs * 128 + c] = d * rstd;
}

// ---------------------------------------------------------------------------
extern "C" void kernel_launch(void* const* d_in, const int* in_sizes, int n_in,
                              void* d_out, int out_size, void* d_ws, size_t ws_size,
                              hipStream_t stream) {
  (void)in_sizes; (void)n_in; (void)out_size; (void)ws_size;
  const float* x   = (const float*)d_in[0];
  const float* gw  = (const float*)d_in[1];
  const float* gb  = (const float*)d_in[2];
  const float* wq  = (const float*)d_in[3];
  const float* bq  = (const float*)d_in[4];
  const float* wk  = (const float*)d_in[5];
  const float* bk  = (const float*)d_in[6];
  const float* c1w = (const float*)d_in[7];
  const float* c1b = (const float*)d_in[8];
  const float* c2w = (const float*)d_in[9];
  const float* c2b = (const float*)d_in[10];
  float* out = (float*)d_out;

  char* ws = (char*)d_ws;
  size_t o = 0;
  auto alloc = [&](size_t bytes) { size_t r = o; o += (bytes + 255) & ~(size_t)255; return r; };

  bf16_t* w1bf = (bf16_t*)(ws + alloc((size_t)512 * 64 * 2));
  bf16_t* wqbf = (bf16_t*)(ws + alloc((size_t)512 * 512 * 2));
  bf16_t* wkbf = (bf16_t*)(ws + alloc((size_t)512 * 512 * 2));
  bf16_t* c1bf = (bf16_t*)(ws + alloc((size_t)DFF_ * 512 * 2));
  bf16_t* c2bf = (bf16_t*)(ws + alloc((size_t)512 * DFF_ * 2));
  bf16_t* h64  = (bf16_t*)(ws + alloc((size_t)G_ * N_ * SC_ * 2));
  bf16_t* Whbf = (bf16_t*)(ws + alloc((size_t)G_ * N_ * 512 * 2));   // later reused as x_bf16
  bf16_t* qbf  = (bf16_t*)(ws + alloc((size_t)G_ * N_ * 512 * 2));   // q+k region reused as h_prime/y (f32)
  bf16_t* kbf  = (bf16_t*)(ws + alloc((size_t)G_ * N_ * 512 * 2));
  float*  scores = (float*)(ws + alloc((size_t)G_ * N_ * N_ * 4));   // reused as gout
  bf16_t* attbf  = (bf16_t*)(ws + alloc((size_t)G_ * N_ * N_ * 2));
  bf16_t* y1bf   = (bf16_t*)(ws + alloc((size_t)32 * N_ * DFF_ * 2)); // 32-batch FFN chunk

  bf16_t* xbf    = Whbf;           // Wh dead after att@Wh
  float*  hprime = (float*)qbf;    // q+k dead after scores
  float*  yf     = hprime;         // h_prime dead after LN1
  float*  gout   = scores;         // scores dead after softmax

  // --- convert operands to bf16 ---
  f2bf_kernel<<<1024, 256, 0, stream>>>(wq, wqbf, 512 * 512);
  f2bf_kernel<<<1024, 256, 0, stream>>>(wk, wkbf, 512 * 512);
  f2bf_kernel<<<4096, 256, 0, stream>>>(c1w, c1bf, DFF_ * 512);
  f2bf_kernel<<<4096, 256, 0, stream>>>(c2w, c2bf, 512 * DFF_);
  w1_pack_kernel<<<128, 256, 0, stream>>>(gw, w1bf);
  h64_pack_kernel<<<4096, 256, 0, stream>>>(x, h64);

  dim3 blk(256);
  // Wh = h64 @ gat_w[:, :64]^T + gat_b   (K=64, bf16 out)
  gemm_wmma_kernel<64, false, false, false, false, true, false>
      <<<dim3(4, G_), blk, 0, stream>>>(h64, w1bf, gb, Whbf, 512, 0);
  // q = Wh @ wq^T + bq ; k = Wh @ wk^T + bk
  gemm_wmma_kernel<512, false, false, false, false, true, false>
      <<<dim3(4, G_), blk, 0, stream>>>(Whbf, wqbf, bq, qbf, 512, 0);
  gemm_wmma_kernel<512, false, false, false, false, true, false>
      <<<dim3(4, G_), blk, 0, stream>>>(Whbf, wkbf, bk, kbf, 512, 0);
  // scores = q @ k^T   (k row-major IS the [N][K] operand)
  gemm_wmma_kernel<512, false, true, false, false, false, false>
      <<<dim3(1, G_), blk, 0, stream>>>(qbf, kbf, nullptr, scores, 128, 0);
  softmax2_kernel<<<G_ * N_, 128, 0, stream>>>(scores, attbf);
  // h_prime = att @ Wh  (Wh is [K=128 n'][N=512 t] row-major -> B_ROWMAJOR)
  gemm_wmma_kernel<128, true, true, false, false, false, false>
      <<<dim3(4, G_), blk, 0, stream>>>(attbf, Whbf, nullptr, hprime, 512, 0);
  ln1_kernel<<<G_ * N_, 256, 0, stream>>>(hprime, x, xbf);
  // FFN, chunked over 4 groups of 32 batches to bound the 2048-wide buffer
  for (int c = 0; c < 4; ++c) {
    gemm_wmma_kernel<512, false, false, false, true, true, true>
        <<<dim3(16, 32), blk, 0, stream>>>(xbf, c1bf, c1b, y1bf, DFF_, c * 32);
    gemm_wmma_kernel<2048, false, false, true, false, false, false>
        <<<dim3(4, 32), blk, 0, stream>>>(y1bf, c2bf, c2b, yf, 512, c * 32);
  }
  ln2_kernel<<<G_ * N_, 256, 0, stream>>>(xbf, yf, gout);
  final_ln_kernel<<<B_ * T_, 128, 0, stream>>>(gout, x, out);
}
